// Attention_13554916786722
// MI455X (gfx1250) — matrix operand order repaired and checked by reading, not compiled, not run on hardware
//
#include <hip/hip_runtime.h>

// Problem constants (match reference): B=4, H=8, S=2048, D=512
#define BB 4
#define HH 8
#define SS 2048
#define DD 512
#define BHN (BB*HH)                 // 32
#define SCALE 0.35355339059327373f  // 1/sqrt(H)

typedef __attribute__((ext_vector_type(8)))  float        v8f;
typedef __attribute__((ext_vector_type(4)))  float        v4f;
typedef __attribute__((ext_vector_type(4)))  unsigned int v4u;
typedef __attribute__((ext_vector_type(16))) __bf16       v16bf;

// LDS tile row stride in halfs: 32 payload + 8 pad -> 80B row stride,
// keeps 16B alignment for b128 reads and breaks 64B-stride bank conflicts.
#define ROWP 40

union FragU { v4u u[2]; v16bf v; };

// ---- fp32 -> packed bf16 (RNE) --------------------------------------------
// Prefer the clang builtin; otherwise emit the gfx1250 HW packed convert
// directly (1 VALU op for 2 elements instead of ~7 for the SW emulation).
__device__ __forceinline__ unsigned pack2(float lo, float hi) {
#if __has_builtin(__builtin_amdgcn_cvt_pk_bf16_f32)
    auto r = __builtin_amdgcn_cvt_pk_bf16_f32(lo, hi);
    return __builtin_bit_cast(unsigned, r);
#else
    unsigned d;
    asm("v_cvt_pk_bf16_f32 %0, %1, %2" : "=v"(d) : "v"(lo), "v"(hi));
    return d;
#endif
}

// ---- CDNA5 async global->LDS (ASYNCcnt-tracked DMA into LDS) --------------
__device__ __forceinline__ unsigned lds_off_u32(const void* p) {
    return (unsigned)(unsigned long long)(const __attribute__((address_space(3))) void*)p;
}
__device__ __forceinline__ void async_copy_b128(unsigned lds_byte, const void* g) {
    asm volatile("global_load_async_to_lds_b128 %0, %1, off"
                 :: "v"(lds_byte), "v"((unsigned long long)g)
                 : "memory");
}
#if __has_builtin(__builtin_amdgcn_s_wait_asynccnt)
#define WAIT_ASYNC0() __builtin_amdgcn_s_wait_asynccnt(0)
#else
#define WAIT_ASYNC0() asm volatile("s_wait_asynccnt 0x0" ::: "memory")
#endif

// ---- register slab for software-pipelined global->LDS staging -------------
struct RegSlab { v4f r[4]; };

// 2 rows (kk, kk+1) x 8 contiguous columns from a (row-major, ld=DD) matrix
__device__ __forceinline__ RegSlab load_rows2x8(const float* base) {
    RegSlab s;
    s.r[0] = *(const v4f*)base;
    s.r[1] = *(const v4f*)(base + 4);
    s.r[2] = *(const v4f*)(base + DD);
    s.r[3] = *(const v4f*)(base + DD + 4);
    return s;
}
// transpose-store into LDS [m][k] layout as bf16 pairs (k even)
__device__ __forceinline__ void store_rows2x8(unsigned short* lds, int mb, int kk,
                                              const RegSlab& s) {
#pragma unroll
    for (int i = 0; i < 4; ++i)
        *(unsigned*)&lds[(mb + i) * ROWP + kk]     = pack2(s.r[0][i], s.r[2][i]);
#pragma unroll
    for (int i = 0; i < 4; ++i)
        *(unsigned*)&lds[(mb + 4 + i) * ROWP + kk] = pack2(s.r[1][i], s.r[3][i]);
}

// ---- one 16(M) x 128(N) x 32(K) WMMA stage: batch-load B frags, then chain
__device__ __forceinline__ void wmma_stage(const unsigned short* la,
                                           const unsigned short* lb,
                                           int wv, int ml, int hl, v8f acc[8]) {
    FragU af;
    const int arow = wv * 16 + ml;
    af.u[0] = *(const v4u*)&la[arow * ROWP + hl * 8];
    af.u[1] = *(const v4u*)&la[arow * ROWP + 16 + hl * 8];
    FragU bf_[8];
#pragma unroll
    for (int nt = 0; nt < 8; ++nt) {
        const int brow = nt * 16 + ml;
        bf_[nt].u[0] = *(const v4u*)&lb[brow * ROWP + hl * 8];
        bf_[nt].u[1] = *(const v4u*)&lb[brow * ROWP + 16 + hl * 8];
    }
#pragma unroll
    for (int nt = 0; nt < 8; ++nt)
        acc[nt] = __builtin_amdgcn_wmma_f32_16x16x32_bf16(
            false, af.v, false, bf_[nt].v, (short)0, acc[nt], false, false);
}

// ---------------------------------------------------------------------------
// Kernel 1: score[bh, d, e] = scale * sum_s q[bh,s,d] * k[bh,s,e]
// 128x128 tile per block; K-slabs of 32 staged through LDS with a
// register-level prefetch pipeline (load slab i+1 while WMMA'ing slab i).
// ---------------------------------------------------------------------------
__global__ __launch_bounds__(256) void qk_kernel(const float* __restrict__ q,
                                                 const float* __restrict__ kmat,
                                                 float* __restrict__ map) {
    __shared__ __align__(16) unsigned short lds_q[128 * ROWP]; // [m][k] bf16
    __shared__ __align__(16) unsigned short lds_k[128 * ROWP]; // [n][k] bf16

    const int bh   = blockIdx.y;
    const int tile = blockIdx.x;            // 0..15
    const int tm   = (tile & 3) * 128;      // d-tile
    const int tn   = (tile >> 2) * 128;     // e-tile
    const int t    = threadIdx.x;
    const int wv   = t >> 5;
    const int lane = t & 31;
    const int hl   = lane >> 4;
    const int ml   = lane & 15;

    const int kk = (t >> 4) * 2;            // staging: k rows {kk, kk+1}
    const int mb = (t & 15) * 8;            // staging: 8 columns

    const float* qp = q    + (size_t)bh * SS * DD + (size_t)kk * DD + tm + mb;
    const float* kp = kmat + (size_t)bh * SS * DD + (size_t)kk * DD + tn + mb;

    v8f acc[8];
#pragma unroll
    for (int i = 0; i < 8; ++i)
#pragma unroll
        for (int j = 0; j < 8; ++j) acc[i][j] = 0.0f;

    RegSlab rq = load_rows2x8(qp);
    RegSlab rk = load_rows2x8(kp);

    for (int s0 = 0; s0 < SS; s0 += 32) {
        store_rows2x8(lds_q, mb, kk, rq);
        store_rows2x8(lds_k, mb, kk, rk);
        __syncthreads();
        if (s0 + 32 < SS) {                  // prefetch next slab into regs
            rq = load_rows2x8(qp + (size_t)(s0 + 32) * DD);
            rk = load_rows2x8(kp + (size_t)(s0 + 32) * DD);
        }
        wmma_stage(lds_q, lds_k, wv, ml, hl, acc);
        __syncthreads();
    }

    // epilogue: scale + store pre-softmax score (atten_map)
    float* mp = map + (size_t)bh * DD * DD;
#pragma unroll
    for (int nt = 0; nt < 8; ++nt) {
        const int col = tn + nt * 16 + ml;
#pragma unroll
        for (int i = 0; i < 8; ++i) {
            const int row = tm + wv * 16 + i + 8 * hl;
            mp[(size_t)row * DD + col] = acc[nt][i] * SCALE;
        }
    }
}

// ---------------------------------------------------------------------------
// Kernel 2: row softmax over 512-wide rows; write bf16 attn into workspace.
// One wave32 per row; each lane owns 16 contiguous columns -> b128 loads,
// packed bf16 b128 stores. Shuffle reductions across the wave.
// ---------------------------------------------------------------------------
__global__ __launch_bounds__(256) void softmax_kernel(const float* __restrict__ map,
                                                      unsigned short* __restrict__ attn) {
    const int lane = threadIdx.x & 31;
    const int wv   = threadIdx.x >> 5;
    const size_t row = (size_t)blockIdx.x * 8 + wv;
    const float* rp = map + row * DD + lane * 16;

    v4f c[4];
#pragma unroll
    for (int j = 0; j < 4; ++j) c[j] = *(const v4f*)(rp + 4 * j);

    float m = -3.402823466e38f;
#pragma unroll
    for (int j = 0; j < 4; ++j)
#pragma unroll
        for (int i = 0; i < 4; ++i) m = fmaxf(m, c[j][i]);
#pragma unroll
    for (int off = 16; off > 0; off >>= 1) m = fmaxf(m, __shfl_xor(m, off, 32));

    float s = 0.0f;
#pragma unroll
    for (int j = 0; j < 4; ++j)
#pragma unroll
        for (int i = 0; i < 4; ++i) { c[j][i] = __expf(c[j][i] - m); s += c[j][i]; }
#pragma unroll
    for (int off = 16; off > 0; off >>= 1) s += __shfl_xor(s, off, 32);
    const float inv = 1.0f / s;

    v4u u0, u1;
    u0[0] = pack2(c[0][0] * inv, c[0][1] * inv);
    u0[1] = pack2(c[0][2] * inv, c[0][3] * inv);
    u0[2] = pack2(c[1][0] * inv, c[1][1] * inv);
    u0[3] = pack2(c[1][2] * inv, c[1][3] * inv);
    u1[0] = pack2(c[2][0] * inv, c[2][1] * inv);
    u1[1] = pack2(c[2][2] * inv, c[2][3] * inv);
    u1[2] = pack2(c[3][0] * inv, c[3][1] * inv);
    u1[3] = pack2(c[3][2] * inv, c[3][3] * inv);

    v4u* op = (v4u*)(attn + row * DD + lane * 16);
    op[0] = u0;
    op[1] = u1;
}

// ---------------------------------------------------------------------------
// Kernel 3: out[bh, d, s] = sum_e attn[bh,d,e] * v[bh,s,e]
// A (attn, bf16) DMA'd into double-buffered LDS via async-to-LDS;
// B (V) converted fp32->bf16 through registers with prefetch pipeline.
// ---------------------------------------------------------------------------
__global__ __launch_bounds__(256) void pv_kernel(const unsigned short* __restrict__ attn,
                                                 const float* __restrict__ vmat,
                                                 float* __restrict__ out) {
    __shared__ __align__(16) unsigned short lds_a[2][128 * ROWP]; // [m][k] bf16
    __shared__ __align__(16) unsigned short lds_b[128 * ROWP];    // [n][k] bf16

    const int bh   = blockIdx.y;
    const int tile = blockIdx.x;            // 0..63
    const int tm   = (tile & 3) * 128;      // d-tile
    const int tn   = (tile >> 2) * 128;     // s-tile
    const int t    = threadIdx.x;
    const int wv   = t >> 5;
    const int lane = t & 31;
    const int hl   = lane >> 4;
    const int ml   = lane & 15;

    const int row2 = t >> 1;   // 0..127
    const int seg  = t & 1;    // 16-half segment of a 32-half row

    const unsigned short* ap = attn + (size_t)bh * DD * DD
                             + (size_t)(tm + row2) * DD + seg * 16;
    const float* vp = vmat + (size_t)bh * SS * DD
                    + (size_t)(tn + row2) * DD + seg * 16;

    const unsigned la0 = lds_off_u32(&lds_a[0][row2 * ROWP + seg * 16]);
    const unsigned la1 = lds_off_u32(&lds_a[1][row2 * ROWP + seg * 16]);

    v8f acc[8];
#pragma unroll
    for (int i = 0; i < 8; ++i)
#pragma unroll
        for (int j = 0; j < 8; ++j) acc[i][j] = 0.0f;

    // prologue: DMA attn slab 0 into buffer 0; V slab 0 into registers
    async_copy_b128(la0,      ap);
    async_copy_b128(la0 + 16, ap + 8);
    v4f f0 = *(const v4f*)vp,       f1 = *(const v4f*)(vp + 4);
    v4f f2 = *(const v4f*)(vp + 8), f3 = *(const v4f*)(vp + 12);

    int it = 0;
    for (int e0 = 0; e0 < DD; e0 += 32, ++it) {
        const int cur = it & 1;
        // convert current V slab into LDS
        {
            v4u u0, u1;
            u0[0] = pack2(f0[0], f0[1]); u0[1] = pack2(f0[2], f0[3]);
            u0[2] = pack2(f1[0], f1[1]); u0[3] = pack2(f1[2], f1[3]);
            u1[0] = pack2(f2[0], f2[1]); u1[1] = pack2(f2[2], f2[3]);
            u1[2] = pack2(f3[0], f3[1]); u1[3] = pack2(f3[2], f3[3]);
            v4u* dst = (v4u*)&lds_b[row2 * ROWP + seg * 16];
            dst[0] = u0; dst[1] = u1;
        }
        WAIT_ASYNC0();                 // attn slab `it` landed in lds_a[cur]
        __syncthreads();
        if (e0 + 32 < DD) {            // kick off next slab's DMA + loads
            const unsigned lan = cur ? la0 : la1;
            async_copy_b128(lan,      ap + (e0 + 32));
            async_copy_b128(lan + 16, ap + (e0 + 32) + 8);
            const float* vn = vp + (e0 + 32);
            f0 = *(const v4f*)vn;       f1 = *(const v4f*)(vn + 4);
            f2 = *(const v4f*)(vn + 8); f3 = *(const v4f*)(vn + 12);
        }
        wmma_stage(lds_a[cur], lds_b, wv, ml, hl, acc);
        __syncthreads();
    }

    float* ob = out + (size_t)bh * DD * SS;
#pragma unroll
    for (int nt = 0; nt < 8; ++nt) {
        const int col = tn + nt * 16 + ml;
#pragma unroll
        for (int i = 0; i < 8; ++i) {
            const int row = tm + wv * 16 + i + 8 * hl;
            ob[(size_t)row * SS + col] = acc[nt][i];
        }
    }
}

// ---------------------------------------------------------------------------
extern "C" void kernel_launch(void* const* d_in, const int* in_sizes, int n_in,
                              void* d_out, int out_size, void* d_ws, size_t ws_size,
                              hipStream_t stream) {
    const float* q = (const float*)d_in[0];
    const float* k = (const float*)d_in[1];
    const float* v = (const float*)d_in[2];

    float* out = (float*)d_out;                         // (B,H,D,S)
    float* map = out + (size_t)BB * HH * DD * SS;       // (B,H,D,D) pre-softmax

    unsigned short* attn_bf = (unsigned short*)d_ws;    // bf16 (B,H,D,D), 16.8 MB

    qk_kernel<<<dim3(16, BHN), 256, 0, stream>>>(q, k, map);
    softmax_kernel<<<(BHN * DD) / 8, 256, 0, stream>>>(map, attn_bf);
    pv_kernel<<<dim3(64, BHN), 256, 0, stream>>>(attn_bf, v, out);
}
